// ScaledDotProductAttention_19310172962905
// MI455X (gfx1250) — matrix operand order repaired and checked by reading, not compiled, not run on hardware
//
#include <hip/hip_runtime.h>

typedef __attribute__((ext_vector_type(16))) _Float16 v16h;
typedef __attribute__((ext_vector_type(8)))  float    v8f;

#define S_LEN 2048
#define DK    64
#define NBH   32                  // B*H
#define QB_PER_BH (S_LEN / 16)    // 128 query blocks of 16 rows
#define WAVES_PER_WG 8
// scale * log2(e): scores kept in exp2 domain, one v_exp_f32 per element
#define C_EXP2 0.18033688011112042f   // 0.125 * 1.4426950408889634
#define T_MASKED (-1.0e6f)            // exp2(-1e6) == 0 exactly in f32
#define T_CLAMP  80.0f                // 2^80 * 2048 << FLT_MAX, no overflow

// ---- cross-lane XOR swizzles (group-of-32 ds_swizzle patterns) ----
__device__ __forceinline__ float swz_xor1(float v){ return __int_as_float(__builtin_amdgcn_ds_swizzle(__float_as_int(v), 0x041F)); }
__device__ __forceinline__ float swz_xor2(float v){ return __int_as_float(__builtin_amdgcn_ds_swizzle(__float_as_int(v), 0x081F)); }
__device__ __forceinline__ float swz_xor4(float v){ return __int_as_float(__builtin_amdgcn_ds_swizzle(__float_as_int(v), 0x101F)); }
__device__ __forceinline__ float swz_xor8(float v){ return __int_as_float(__builtin_amdgcn_ds_swizzle(__float_as_int(v), 0x201F)); }

__device__ __forceinline__ void cvt8(v16h& a, int base, float4 u, float4 v) {
  a[base+0]=(_Float16)u.x; a[base+1]=(_Float16)u.y; a[base+2]=(_Float16)u.z; a[base+3]=(_Float16)u.w;
  a[base+4]=(_Float16)v.x; a[base+5]=(_Float16)v.y; a[base+6]=(_Float16)v.z; a[base+7]=(_Float16)v.w;
}

// B-fragment of K^T for one 16-key tile, d-chunk c (d = c*32 .. c*32+31).
// Lane l: column N = l&15 (key n0+N); K-dim = c*32 + (l>>4)*8 + i (i<8), c*32 + 16 + (l>>4)*8 + (i-8) (i>=8).
__device__ __forceinline__ v16h loadKfrag(const float* __restrict__ Kb, int n0, int col, int half, int c) {
  const float4* q = (const float4*)(Kb + (size_t)(n0 + col) * DK + c * 32 + half * 8);
  v16h b;
  cvt8(b, 0, q[0], q[1]);
  cvt8(b, 8, q[4], q[5]);
  return b;
}

// 16x16 tile of Q K^T (queries q0..q0+15 x keys n0..n0+15), f32 accumulate.
__device__ __forceinline__ v8f scoreTile(const v16h aq0, const v16h aq1,
                                         const float* __restrict__ Kb,
                                         int n0, int col, int half) {
  v8f acc = {};
  acc = __builtin_amdgcn_wmma_f32_16x16x32_f16(false, aq0, false, loadKfrag(Kb, n0, col, half, 0),
                                               (short)0, acc, false, false);
  acc = __builtin_amdgcn_wmma_f32_16x16x32_f16(false, aq1, false, loadKfrag(Kb, n0, col, half, 1),
                                               (short)0, acc, false, false);
  return acc;
}

// t = clamped exp2-domain score for row (rb+j), col (n0+col)
__device__ __forceinline__ float tval(float accj, unsigned char mbyte) {
  float t = accj * C_EXP2;
  if (mbyte) t = T_MASKED;
  return fminf(t, T_CLAMP);
}

__global__ __launch_bounds__(WAVES_PER_WG * 32)
void sdpa_fused_kernel(const float* __restrict__ Q, const float* __restrict__ K,
                       const float* __restrict__ V, const unsigned char* __restrict__ M,
                       float* __restrict__ ctx_out, float* __restrict__ sc_out) {
  __shared__ _Float16 pstage[WAVES_PER_WG][16][34];   // P relayout staging (+pad vs bank conflicts)

  const int lane = threadIdx.x & 31;
  const int w    = threadIdx.x >> 5;
  const int gid  = blockIdx.x * WAVES_PER_WG + w;     // query-block id
  const int bh   = gid >> 7;                          // / QB_PER_BH
  const int q0   = (gid & (QB_PER_BH - 1)) * 16;
  const int col  = lane & 15;                         // B/C/D column, A row
  const int half = lane >> 4;
  const int rb   = half * 8;                          // C/D row base for this lane

  const float*         Qb = Q + (size_t)bh * S_LEN * DK;
  const float*         Kb = K + (size_t)bh * S_LEN * DK;
  const float*         Vb = V + (size_t)bh * S_LEN * DK;
  const unsigned char* Mb = M + (size_t)bh * S_LEN * S_LEN;
  float*               Ob = ctx_out + (size_t)bh * S_LEN * DK;
  float*               Sb = sc_out  + (size_t)bh * S_LEN * S_LEN;

  // ---- A-fragments of Q (16x32 each, d-chunks 0 and 1) ----
  v16h aq0, aq1;
  {
    const float* qrow = Qb + (size_t)(q0 + col) * DK;
    const float4* p0 = (const float4*)(qrow + half * 8);
    const float4* p1 = (const float4*)(qrow + 16 + half * 8);
    cvt8(aq0, 0, p0[0], p0[1]); cvt8(aq0, 8, p1[0], p1[1]);
    const float4* p2 = (const float4*)(qrow + 32 + half * 8);
    const float4* p3 = (const float4*)(qrow + 48 + half * 8);
    cvt8(aq1, 0, p2[0], p2[1]); cvt8(aq1, 8, p3[0], p3[1]);
  }

  // ---- Pass 1: per-lane sum of 2^t over all keys (shift-free softmax) ----
  float rsum[8];
#pragma unroll
  for (int j = 0; j < 8; ++j) rsum[j] = 0.0f;

  for (int n0 = 0; n0 < S_LEN; n0 += 16) {
    if (n0 + 16 < S_LEN)
      __builtin_prefetch(Kb + (size_t)(n0 + 16 + col) * DK, 0, 3);
    v8f acc = scoreTile(aq0, aq1, Kb, n0, col, half);
    const unsigned char* mrow = Mb + (size_t)(q0 + rb) * S_LEN + n0 + col;
#pragma unroll
    for (int j = 0; j < 8; ++j) {
      float t = tval(acc[j], mrow[(size_t)j * S_LEN]);
      rsum[j] += __builtin_amdgcn_exp2f(t);
    }
  }

  // ---- combine sums across the 16 lanes holding each row ----
  float rinv[8];
#pragma unroll
  for (int j = 0; j < 8; ++j) {
    float s = rsum[j];
    s += swz_xor1(s);
    s += swz_xor2(s);
    s += swz_xor4(s);
    s += swz_xor8(s);
    rinv[j] = 1.0f / fmaxf(s, 1.0e-30f);
  }

  // ---- Pass 2: recompute scores, write normalized P once, accumulate P·V ----
  v8f ctxa[4];
#pragma unroll
  for (int g = 0; g < 4; ++g) ctxa[g] = (v8f){};

  for (int kg = 0; kg < S_LEN; kg += 32) {
#pragma unroll
    for (int t = 0; t < 2; ++t) {
      const int n0 = kg + t * 16;
      v8f acc = scoreTile(aq0, aq1, Kb, n0, col, half);
      const unsigned char* mrow = Mb + (size_t)(q0 + rb) * S_LEN + n0 + col;
      float* srow = Sb + (size_t)(q0 + rb) * S_LEN + n0 + col;
#pragma unroll
      for (int j = 0; j < 8; ++j) {
        float tv = tval(acc[j], mrow[(size_t)j * S_LEN]);
        float p  = __builtin_amdgcn_exp2f(tv) * rinv[j];
        srow[(size_t)j * S_LEN] = p;                       // coalesced 64B/row
        pstage[w][rb + j][t * 16 + col] = (_Float16)p;     // C-layout -> LDS
      }
    }
    asm volatile("s_wait_dscnt 0" ::: "memory");           // same-wave LDS RAW

    // A-fragment of P (16 queries x 32 keys) from LDS
    v16h ap;
    {
      const _Float16* prow = &pstage[w][col][0];
#pragma unroll
      for (int j = 0; j < 8; ++j) {
        ap[j]     = prow[half * 8 + j];
        ap[8 + j] = prow[16 + half * 8 + j];
      }
    }

    // B-fragments of V (32 keys x 16 d-cols per group); lanes coalesce per element
#pragma unroll
    for (int g = 0; g < 4; ++g) {
      v16h bv;
      const float* vcol = Vb + (size_t)(kg + half * 8) * DK + g * 16 + col;
#pragma unroll
      for (int i = 0; i < 16; ++i) bv[i] = (_Float16)vcol[(size_t)(i + (i & 8)) * DK];
      ctxa[g] = __builtin_amdgcn_wmma_f32_16x16x32_f16(false, ap, false, bv,
                                                       (short)0, ctxa[g], false, false);
    }
  }

  // ---- write context ----
#pragma unroll
  for (int g = 0; g < 4; ++g)
#pragma unroll
    for (int j = 0; j < 8; ++j)
      Ob[(size_t)(q0 + rb + j) * DK + g * 16 + col] = ctxa[g][j];
}

extern "C" void kernel_launch(void* const* d_in, const int* in_sizes, int n_in,
                              void* d_out, int out_size, void* d_ws, size_t ws_size,
                              hipStream_t stream) {
  (void)in_sizes; (void)n_in; (void)out_size; (void)d_ws; (void)ws_size;
  const float*         Q = (const float*)d_in[0];
  const float*         K = (const float*)d_in[1];
  const float*         V = (const float*)d_in[2];
  const unsigned char* M = (const unsigned char*)d_in[3];   // jnp bool = 1 byte
  float* ctx = (float*)d_out;                                // [B,H,S,D] first
  float* sc  = (float*)d_out + (size_t)NBH * S_LEN * DK;     // then [B,H,S,S]

  dim3 grid(NBH * QB_PER_BH / WAVES_PER_WG);                 // 512 workgroups
  dim3 block(WAVES_PER_WG * 32);                             // 8 waves (wave32)
  sdpa_fused_kernel<<<grid, block, 0, stream>>>(Q, K, V, M, ctx, sc);
}
